// NCAStep_76673756168967
// MI455X (gfx1250) — matrix operand order repaired and checked by reading, not compiled, not run on hardware
//
#include <hip/hip_runtime.h>
#include <hip/hip_bf16.h>
#include <stddef.h>

typedef __bf16 bf16;
typedef __attribute__((ext_vector_type(16))) __bf16 v16bf;
typedef __attribute__((ext_vector_type(8)))  __bf16 v8bf;
typedef __attribute__((ext_vector_type(8)))  float  v8f;

#define BQ   4
#define LQ   4096
#define DQ   1024
#define BLQ  (BQ * LQ)   // 16384 tokens
#define DIQ  2048        // expanded dim
#define K3D  3072        // 3*D
#define TWOD 2048        // 2*D

// ---- gfx1250 async direct-to-LDS path (guarded; falls back to ds_store) ----
#if defined(__gfx1250__) && __has_builtin(__builtin_amdgcn_global_load_async_to_lds_b128)
#define HAS_ASYNC 1
#else
#define HAS_ASYNC 0
#endif

#if HAS_ASYNC
#if __has_builtin(__builtin_amdgcn_s_wait_asynccnt)
#define WAIT_ASYNC(n) __builtin_amdgcn_s_wait_asynccnt(n)
#else
#define WAIT_ASYNC(n) asm volatile("s_wait_asynccnt %0" ::"n"(n) : "memory")
#endif
typedef int v4i_ __attribute__((vector_size(4 * sizeof(int))));
typedef __attribute__((address_space(1))) v4i_ gv4i;
typedef __attribute__((address_space(3))) v4i_ lv4i;
__device__ __forceinline__ void async_ld_b128(const bf16* g, bf16* l) {
    __builtin_amdgcn_global_load_async_to_lds_b128((gv4i*)g, (lv4i*)l, 0, 0);
}
#endif

__device__ __forceinline__ float sigm(float x) { return 1.0f / (1.0f + __expf(-x)); }

// ---------------------------------------------------------------------------
// f32 -> bf16 conversion (weights)
// ---------------------------------------------------------------------------
__global__ void cvt_f32_bf16(const float* __restrict__ s, bf16* __restrict__ d, int n) {
    int i = blockIdx.x * blockDim.x + threadIdx.x;
    if (i < n) d[i] = (bf16)s[i];
}

// ---------------------------------------------------------------------------
// PerceptionFilter depthwise convs (k=7, dil=1) -> pcat bf16 [BL, 3D]
// also writes x (bf16) into cat[:, 0:D]
// ---------------------------------------------------------------------------
__global__ __launch_bounds__(256)
void prep_kernel(const float* __restrict__ X,
                 const float* __restrict__ w0, const float* __restrict__ w1,
                 const float* __restrict__ w2,
                 bf16* __restrict__ pcat, bf16* __restrict__ cat) {
    int idx = blockIdx.x * 256 + threadIdx.x;   // over BL*D
    int d   = idx & (DQ - 1);
    int row = idx >> 10;
    int l   = row & (LQ - 1);
    float v[7];
#pragma unroll
    for (int j = 0; j < 7; ++j) {
        int lj = l + j - 3;
        v[j] = (lj >= 0 && lj < LQ) ? X[(size_t)(row + j - 3) * DQ + d] : 0.0f;
    }
    float a0 = 0.f, a1 = 0.f, a2 = 0.f;
#pragma unroll
    for (int j = 0; j < 7; ++j) {
        a0 += v[j] * w0[d * 7 + j];
        a1 += v[j] * w1[d * 7 + j];
        a2 += v[j] * w2[d * 7 + j];
    }
    size_t pr = (size_t)row * K3D;
    pcat[pr + d]          = (bf16)a0;
    pcat[pr + DQ + d]     = (bf16)a1;
    pcat[pr + 2 * DQ + d] = (bf16)a2;
    cat[(size_t)row * TWOD + d] = (bf16)v[3];
}

// ---------------------------------------------------------------------------
// WMMA GEMM: C[M,N] = A[M,K] * Bw[N,K]^T   (bf16 in, f32 acc)
// EPI==0: C = acc (f32)        EPI==1: C = xres + alpha[col]*acc (f32)
// Block tile 128x128x32, 8 waves (2x4), wave tile 64x32 = 4x2 WMMA tiles.
// Async double-buffered LDS staging when available.
// ---------------------------------------------------------------------------
template <int EPI>
__global__ __launch_bounds__(256)
void gemm_nt(const bf16* __restrict__ A, const bf16* __restrict__ Bw,
             float* __restrict__ C, const float* __restrict__ xres,
             const float* __restrict__ alpha, int M, int N, int K) {
#if HAS_ASYNC
    __shared__ __align__(16) bf16 As[2][128][40];
    __shared__ __align__(16) bf16 Bs[2][128][40];
#else
    __shared__ __align__(16) bf16 As[1][128][40];
    __shared__ __align__(16) bf16 Bs[1][128][40];
#endif
    const int tid  = threadIdx.x;
    const int lane = tid & 31;
    const int wave = tid >> 5;
    const int wm   = wave >> 2;   // 0..1
    const int wn   = wave & 3;    // 0..3
    const int rowBase = blockIdx.y * 128;
    const int colBase = blockIdx.x * 128;

    // staging: two 16B chunks per thread per matrix (rows r0 and r0+64)
    const int r0 = tid >> 2;             // 0..63
    const int c0 = (tid & 3) * 8;        // 0,8,16,24
    const bf16* gA0 = A  + (size_t)(rowBase + r0) * K + c0;
    const bf16* gA1 = gA0 + (size_t)64 * K;
    const bf16* gB0 = Bw + (size_t)(colBase + r0) * K + c0;
    const bf16* gB1 = gB0 + (size_t)64 * K;

    v8f acc[4][2];
#pragma unroll
    for (int mi = 0; mi < 4; ++mi)
#pragma unroll
        for (int ni = 0; ni < 2; ++ni)
#pragma unroll
            for (int r = 0; r < 8; ++r) acc[mi][ni][r] = 0.0f;

    const int ml = lane & 15;
    const int kb = (lane >> 4) * 8;      // A fragment k-base
    const int kh = (lane >> 4) * 16;     // unused; B fragment also uses kb

    auto compute = [&](const bf16 (*Asl)[40], const bf16 (*Bsl)[40]) {
        v16bf af[4];
#pragma unroll
        for (int mi = 0; mi < 4; ++mi) {
            int m = wm * 64 + mi * 16 + ml;
            v8bf lo = *reinterpret_cast<const v8bf*>(&Asl[m][kb]);
            v8bf hi = *reinterpret_cast<const v8bf*>(&Asl[m][kb + 16]);
#pragma unroll
            for (int e = 0; e < 8; ++e) { af[mi][e] = lo[e]; af[mi][e + 8] = hi[e]; }
        }
        v16bf bfg[2];
#pragma unroll
        for (int ni = 0; ni < 2; ++ni) {
            int n = wn * 32 + ni * 16 + ml;
            v8bf lo = *reinterpret_cast<const v8bf*>(&Bsl[n][kb]);
            v8bf hi = *reinterpret_cast<const v8bf*>(&Bsl[n][kb + 16]);
#pragma unroll
            for (int e = 0; e < 8; ++e) { bfg[ni][e] = lo[e]; bfg[ni][e + 8] = hi[e]; }
        }
#pragma unroll
        for (int mi = 0; mi < 4; ++mi)
#pragma unroll
            for (int ni = 0; ni < 2; ++ni)
                acc[mi][ni] = __builtin_amdgcn_wmma_f32_16x16x32_bf16(
                    false, af[mi], false, bfg[ni], (short)0, acc[mi][ni], false, false);
    };

    const int NK = K >> 5;
#if HAS_ASYNC
    // prologue: tile 0 -> buffer 0
    async_ld_b128(gA0, &As[0][r0][c0]);
    async_ld_b128(gA1, &As[0][r0 + 64][c0]);
    async_ld_b128(gB0, &Bs[0][r0][c0]);
    async_ld_b128(gB1, &Bs[0][r0 + 64][c0]);
    for (int kt = 0; kt < NK; ++kt) {
        const int buf = kt & 1;
        __syncthreads();                       // buf^1 free for overwrite
        if (kt + 1 < NK) {
            const int k0 = (kt + 1) << 5;
            async_ld_b128(gA0 + k0, &As[buf ^ 1][r0][c0]);
            async_ld_b128(gA1 + k0, &As[buf ^ 1][r0 + 64][c0]);
            async_ld_b128(gB0 + k0, &Bs[buf ^ 1][r0][c0]);
            async_ld_b128(gB1 + k0, &Bs[buf ^ 1][r0 + 64][c0]);
            WAIT_ASYNC(4);                     // oldest 4 (tile kt) complete
        } else {
            WAIT_ASYNC(0);
        }
        __syncthreads();                       // tile kt visible to all waves
        compute(As[buf], Bs[buf]);
    }
#else
    for (int kt = 0; kt < NK; ++kt) {
        const int k0 = kt << 5;
        __syncthreads();
        *reinterpret_cast<uint4*>(&As[0][r0][c0])      = *reinterpret_cast<const uint4*>(gA0 + k0);
        *reinterpret_cast<uint4*>(&As[0][r0 + 64][c0]) = *reinterpret_cast<const uint4*>(gA1 + k0);
        *reinterpret_cast<uint4*>(&Bs[0][r0][c0])      = *reinterpret_cast<const uint4*>(gB0 + k0);
        *reinterpret_cast<uint4*>(&Bs[0][r0 + 64][c0]) = *reinterpret_cast<const uint4*>(gB1 + k0);
        __syncthreads();
        compute(As[0], Bs[0]);
    }
#endif

    const int nIn = lane & 15;
    const int mHi = (lane >> 4) * 8;
#pragma unroll
    for (int mi = 0; mi < 4; ++mi)
#pragma unroll
        for (int ni = 0; ni < 2; ++ni) {
            int col = colBase + wn * 32 + ni * 16 + nIn;
            int rb  = rowBase + wm * 64 + mi * 16 + mHi;
#pragma unroll
            for (int r = 0; r < 8; ++r) {
                size_t off = (size_t)(rb + r) * N + col;
                float v = acc[mi][ni][r];
                if (EPI == 0) C[off] = v;
                else          C[off] = xres[off] + alpha[col] * v;
            }
        }
}

// ---------------------------------------------------------------------------
// LayerNorm per token over D, writes p (bf16) into cat[:, D:2D]
// ---------------------------------------------------------------------------
__global__ __launch_bounds__(256)
void ln_kernel(const float* __restrict__ P, const float* __restrict__ g,
               const float* __restrict__ bta, bf16* __restrict__ cat) {
    __shared__ float red[256];
    int row = blockIdx.x, tid = threadIdx.x;
    float v[4]; float s = 0.f;
#pragma unroll
    for (int i = 0; i < 4; ++i) {
        int d = tid + i * 256;
        v[i] = P[(size_t)row * DQ + d];
        s += v[i];
    }
    red[tid] = s; __syncthreads();
    for (int st = 128; st > 0; st >>= 1) { if (tid < st) red[tid] += red[tid + st]; __syncthreads(); }
    float mean = red[0] * (1.0f / DQ);
    __syncthreads();
    float s2 = 0.f;
#pragma unroll
    for (int i = 0; i < 4; ++i) { float dd = v[i] - mean; s2 += dd * dd; }
    red[tid] = s2; __syncthreads();
    for (int st = 128; st > 0; st >>= 1) { if (tid < st) red[tid] += red[tid + st]; __syncthreads(); }
    float rstd = rsqrtf(red[0] * (1.0f / DQ) + 1e-5f);
#pragma unroll
    for (int i = 0; i < 4; ++i) {
        int d = tid + i * 256;
        float o = (v[i] - mean) * rstd * g[d] + bta[d];
        cat[(size_t)row * TWOD + DQ + d] = (bf16)o;
    }
}

// ---------------------------------------------------------------------------
// Dual-output WMMA GEMM: U = cat*Wu^T, G = cat*Wg^T, fused h = sigmoid(G)*silu(U)
// Block tile 128x64x32; 8 waves (2x4), wave tile 64x16. K = N = 2048.
// ---------------------------------------------------------------------------
__global__ __launch_bounds__(256)
void gemm_ug(const bf16* __restrict__ A, const bf16* __restrict__ Wu,
             const bf16* __restrict__ Wg, bf16* __restrict__ H) {
    const int K = TWOD, N = DIQ;
#if HAS_ASYNC
    __shared__ __align__(16) bf16 As[2][128][40];
    __shared__ __align__(16) bf16 Bu[2][64][40];
    __shared__ __align__(16) bf16 Bg[2][64][40];
#else
    __shared__ __align__(16) bf16 As[1][128][40];
    __shared__ __align__(16) bf16 Bu[1][64][40];
    __shared__ __align__(16) bf16 Bg[1][64][40];
#endif
    const int tid  = threadIdx.x;
    const int lane = tid & 31;
    const int wave = tid >> 5;
    const int wm   = wave >> 2;
    const int wn   = wave & 3;
    const int rowBase = blockIdx.y * 128;
    const int colBase = blockIdx.x * 64;

    const int r0 = tid >> 2;        // 0..63
    const int c0 = (tid & 3) * 8;
    const bf16* gA0 = A  + (size_t)(rowBase + r0) * K + c0;
    const bf16* gA1 = gA0 + (size_t)64 * K;
    const bf16* gU  = Wu + (size_t)(colBase + r0) * K + c0;
    const bf16* gG  = Wg + (size_t)(colBase + r0) * K + c0;

    v8f accU[4], accG[4];
#pragma unroll
    for (int mi = 0; mi < 4; ++mi)
#pragma unroll
        for (int r = 0; r < 8; ++r) { accU[mi][r] = 0.0f; accG[mi][r] = 0.0f; }

    const int ml = lane & 15;
    const int kb = (lane >> 4) * 8;
    const int kh = (lane >> 4) * 16;
    const int nB = wn * 16 + ml;

    auto compute = [&](const bf16 (*Asl)[40], const bf16 (*Bul)[40], const bf16 (*Bgl)[40]) {
        v16bf af[4];
#pragma unroll
        for (int mi = 0; mi < 4; ++mi) {
            int m = wm * 64 + mi * 16 + ml;
            v8bf lo = *reinterpret_cast<const v8bf*>(&Asl[m][kb]);
            v8bf hi = *reinterpret_cast<const v8bf*>(&Asl[m][kb + 16]);
#pragma unroll
            for (int e = 0; e < 8; ++e) { af[mi][e] = lo[e]; af[mi][e + 8] = hi[e]; }
        }
        v16bf bu, bg;
        {
            v8bf lo = *reinterpret_cast<const v8bf*>(&Bul[nB][kb]);
            v8bf hi = *reinterpret_cast<const v8bf*>(&Bul[nB][kb + 16]);
#pragma unroll
            for (int e = 0; e < 8; ++e) { bu[e] = lo[e]; bu[e + 8] = hi[e]; }
        }
        {
            v8bf lo = *reinterpret_cast<const v8bf*>(&Bgl[nB][kb]);
            v8bf hi = *reinterpret_cast<const v8bf*>(&Bgl[nB][kb + 16]);
#pragma unroll
            for (int e = 0; e < 8; ++e) { bg[e] = lo[e]; bg[e + 8] = hi[e]; }
        }
#pragma unroll
        for (int mi = 0; mi < 4; ++mi) {
            accU[mi] = __builtin_amdgcn_wmma_f32_16x16x32_bf16(
                false, af[mi], false, bu, (short)0, accU[mi], false, false);
            accG[mi] = __builtin_amdgcn_wmma_f32_16x16x32_bf16(
                false, af[mi], false, bg, (short)0, accG[mi], false, false);
        }
    };

    const int NK = K >> 5;
#if HAS_ASYNC
    async_ld_b128(gA0, &As[0][r0][c0]);
    async_ld_b128(gA1, &As[0][r0 + 64][c0]);
    async_ld_b128(gU,  &Bu[0][r0][c0]);
    async_ld_b128(gG,  &Bg[0][r0][c0]);
    for (int kt = 0; kt < NK; ++kt) {
        const int buf = kt & 1;
        __syncthreads();
        if (kt + 1 < NK) {
            const int k0 = (kt + 1) << 5;
            async_ld_b128(gA0 + k0, &As[buf ^ 1][r0][c0]);
            async_ld_b128(gA1 + k0, &As[buf ^ 1][r0 + 64][c0]);
            async_ld_b128(gU  + k0, &Bu[buf ^ 1][r0][c0]);
            async_ld_b128(gG  + k0, &Bg[buf ^ 1][r0][c0]);
            WAIT_ASYNC(4);
        } else {
            WAIT_ASYNC(0);
        }
        __syncthreads();
        compute(As[buf], Bu[buf], Bg[buf]);
    }
#else
    for (int kt = 0; kt < NK; ++kt) {
        const int k0 = kt << 5;
        __syncthreads();
        *reinterpret_cast<uint4*>(&As[0][r0][c0])      = *reinterpret_cast<const uint4*>(gA0 + k0);
        *reinterpret_cast<uint4*>(&As[0][r0 + 64][c0]) = *reinterpret_cast<const uint4*>(gA1 + k0);
        *reinterpret_cast<uint4*>(&Bu[0][r0][c0])      = *reinterpret_cast<const uint4*>(gU + k0);
        *reinterpret_cast<uint4*>(&Bg[0][r0][c0])      = *reinterpret_cast<const uint4*>(gG + k0);
        __syncthreads();
        compute(As[0], Bu[0], Bg[0]);
    }
#endif

    const int nIn = lane & 15;
    const int mHi = (lane >> 4) * 8;
    int col = colBase + wn * 16 + nIn;
#pragma unroll
    for (int mi = 0; mi < 4; ++mi) {
        int rb = rowBase + wm * 64 + mi * 16 + mHi;
#pragma unroll
        for (int r = 0; r < 8; ++r) {
            float u = accU[mi][r], g = accG[mi][r];
            float hv = sigm(g) * u * sigm(u);       // sigmoid(g) * silu(u)
            H[(size_t)(rb + r) * N + col] = (bf16)hv;
        }
    }
}

// ---------------------------------------------------------------------------
// MultiRateDiffusion: selector softmax + 3 dilated k=3 depthwise convs + blend
// One block per token.
// ---------------------------------------------------------------------------
__global__ __launch_bounds__(256)
void diffusion_kernel(const float* __restrict__ X2,
                      const float* __restrict__ w0, const float* __restrict__ w1,
                      const float* __restrict__ w2,
                      const float* __restrict__ sel, const float* __restrict__ strength,
                      float* __restrict__ out) {
    __shared__ float red[3][256];
    int row = blockIdx.x;
    int l   = row & (LQ - 1);
    int tid = threadIdx.x;
    float xc[4]; float s0 = 0.f, s1 = 0.f, s2 = 0.f;
#pragma unroll
    for (int i = 0; i < 4; ++i) {
        int d = tid + i * 256;
        float v = X2[(size_t)row * DQ + d];
        xc[i] = v;
        s0 += v * sel[d];
        s1 += v * sel[DQ + d];
        s2 += v * sel[2 * DQ + d];
    }
    red[0][tid] = s0; red[1][tid] = s1; red[2][tid] = s2; __syncthreads();
    for (int st = 128; st > 0; st >>= 1) {
        if (tid < st) {
            red[0][tid] += red[0][tid + st];
            red[1][tid] += red[1][tid + st];
            red[2][tid] += red[2][tid + st];
        }
        __syncthreads();
    }
    float L0 = red[0][0], L1 = red[1][0], L2 = red[2][0];
    float mx = fmaxf(L0, fmaxf(L1, L2));
    float e0 = __expf(L0 - mx), e1 = __expf(L1 - mx), e2 = __expf(L2 - mx);
    float inv = 1.0f / (e0 + e1 + e2);
    float ws0 = e0 * inv, ws1 = e1 * inv, ws2 = e2 * inv;
    float str = strength[0];
#pragma unroll
    for (int i = 0; i < 4; ++i) {
        int d = tid + i * 256;
        float c = xc[i];
        float comb = 0.f;
        {   // dil = 1
            float lf = (l >= 1)      ? X2[(size_t)(row - 1) * DQ + d]  : 0.f;
            float rt = (l + 1 < LQ)  ? X2[(size_t)(row + 1) * DQ + d]  : 0.f;
            comb += ws0 * (lf * w0[d * 3 + 0] + c * w0[d * 3 + 1] + rt * w0[d * 3 + 2]);
        }
        {   // dil = 4
            float lf = (l >= 4)      ? X2[(size_t)(row - 4) * DQ + d]  : 0.f;
            float rt = (l + 4 < LQ)  ? X2[(size_t)(row + 4) * DQ + d]  : 0.f;
            comb += ws1 * (lf * w1[d * 3 + 0] + c * w1[d * 3 + 1] + rt * w1[d * 3 + 2]);
        }
        {   // dil = 16
            float lf = (l >= 16)     ? X2[(size_t)(row - 16) * DQ + d] : 0.f;
            float rt = (l + 16 < LQ) ? X2[(size_t)(row + 16) * DQ + d] : 0.f;
            comb += ws2 * (lf * w2[d * 3 + 0] + c * w2[d * 3 + 1] + rt * w2[d * 3 + 2]);
        }
        out[(size_t)row * DQ + d] = c + str * (comb - c);
    }
}

// ---------------------------------------------------------------------------
extern "C" void kernel_launch(void* const* d_in, const int* in_sizes, int n_in,
                              void* d_out, int out_size, void* d_ws, size_t ws_size,
                              hipStream_t stream) {
    (void)in_sizes; (void)n_in; (void)out_size; (void)ws_size;
    const float* x      = (const float*)d_in[0];
    const float* pf_w0  = (const float*)d_in[1];
    const float* pf_w1  = (const float*)d_in[2];
    const float* pf_w2  = (const float*)d_in[3];
    const float* pf_pw  = (const float*)d_in[4];
    const float* ln_g   = (const float*)d_in[5];
    const float* ln_b   = (const float*)d_in[6];
    const float* rg_wu  = (const float*)d_in[7];
    const float* rg_wg  = (const float*)d_in[8];
    const float* rg_wd  = (const float*)d_in[9];
    const float* rg_al  = (const float*)d_in[10];
    const float* md_w0  = (const float*)d_in[11];
    const float* md_w1  = (const float*)d_in[12];
    const float* md_w2  = (const float*)d_in[13];
    const float* md_sel = (const float*)d_in[14];
    const float* md_str = (const float*)d_in[15];
    float* out = (float*)d_out;

    char* ws = (char*)d_ws;
    bf16*  pcat  = (bf16*)(ws + 0);            // [BL,3072] bf16 (96MB) -> reused as h [BL,2048]
    bf16*  cat   = (bf16*)(ws + 100663296);    // [BL,2048] bf16 (64MB): [x | p]
    float* praw  = (float*)(ws + 167772160);   // [BL,1024] f32 (64MB) -> reused as x2
    bf16*  wpw_b = (bf16*)(ws + 234881024);    // pf_pw bf16  (6MB)
    bf16*  wu_b  = (bf16*)(ws + 241172480);    // rg_wu bf16  (8MB)
    bf16*  wg_b  = (bf16*)(ws + 249561088);    // rg_wg bf16  (8MB)
    bf16*  wd_b  = (bf16*)(ws + 257949696);    // rg_wd bf16  (4MB)
    bf16*  h     = pcat;
    float* x2    = praw;

    // weight downcasts
    cvt_f32_bf16<<<(K3D * DQ + 255) / 256, 256, 0, stream>>>(pf_pw, wpw_b, K3D * DQ);
    cvt_f32_bf16<<<(DIQ * TWOD + 255) / 256, 256, 0, stream>>>(rg_wu, wu_b, DIQ * TWOD);
    cvt_f32_bf16<<<(DIQ * TWOD + 255) / 256, 256, 0, stream>>>(rg_wg, wg_b, DIQ * TWOD);
    cvt_f32_bf16<<<(DQ * DIQ + 255) / 256, 256, 0, stream>>>(rg_wd, wd_b, DQ * DIQ);

    // PerceptionFilter depthwise convs + x into cat
    prep_kernel<<<(BLQ * DQ) / 256, 256, 0, stream>>>(x, pf_w0, pf_w1, pf_w2, pcat, cat);

    // GEMM1: p_raw = pcat @ pf_pw^T
    gemm_nt<0><<<dim3(DQ / 128, BLQ / 128), 256, 0, stream>>>(
        pcat, wpw_b, praw, nullptr, nullptr, BLQ, DQ, K3D);

    // LayerNorm -> cat[:, D:]
    ln_kernel<<<BLQ, 256, 0, stream>>>(praw, ln_g, ln_b, cat);

    // GEMM2: h = sigmoid(cat@Wg^T) * silu(cat@Wu^T)   (overwrites pcat region)
    gemm_ug<<<dim3(DIQ / 64, BLQ / 128), 256, 0, stream>>>(cat, wu_b, wg_b, h);

    // GEMM3: x2 = x + alpha * (h @ rg_wd^T)           (overwrites praw region)
    gemm_nt<1><<<dim3(DQ / 128, BLQ / 128), 256, 0, stream>>>(
        h, wd_b, x2, x, rg_al, BLQ, DQ, DIQ);

    // MultiRateDiffusion + final blend
    diffusion_kernel<<<BLQ, 256, 0, stream>>>(x2, md_w0, md_w1, md_w2, md_sel, md_str, out);
}
